// SelfAttention_9801115369787
// MI455X (gfx1250) — matrix operand order repaired and checked by reading, not compiled, not run on hardware
//
#include <hip/hip_runtime.h>
#include <hip/hip_bf16.h>

typedef __attribute__((ext_vector_type(16))) _Float16 v16h;
typedef __attribute__((ext_vector_type(8)))  _Float16 v8h;
typedef __attribute__((ext_vector_type(8)))  float    v8f;
typedef __attribute__((ext_vector_type(4)))  float    v4f;

#define DIM   256
#define BATCH 8
#define SEQ   2048
#define STOT  (BATCH*SEQ)
#define QKSCALE 0.0625f              // 1/sqrt(256)
#define TILES  ((STOT/16)*(DIM/16))  // 16384 output tiles per [STOT x DIM] matrix

// ---------------- WMMA helpers ----------------

__device__ __forceinline__ v8f wmma16(v16h a, v16h b, v8f c) {
  // D = A(16x32 f16) * B(32x16 f16) + C(16x16 f32)
  return __builtin_amdgcn_wmma_f32_16x16x32_f16(
      /*neg_a=*/false, a, /*neg_b=*/false, b,
      /*c_mod=*/(short)0, c, /*reuse_a=*/false, /*reuse_b=*/false);
}

// A-fragment (16x32, f16), row-major source. rowp points at this lane's row.
// lanes 0-15: elems[0..7]=K0..7, elems[8..15]=K16..23
// lanes16-31: elems[0..7]=K8..15, elems[8..15]=K24..31
__device__ __forceinline__ v16h load_afrag_f16(const _Float16* rowp, int k0, int ln) {
  const _Float16* p = rowp + k0 + ((ln >> 4) << 3);
  v8h lo = *(const v8h*)(p);
  v8h hi = *(const v8h*)(p + 16);
  v16h r;
  #pragma unroll
  for (int i = 0; i < 8; ++i) { r[i] = lo[i]; r[i + 8] = hi[i]; }
  return r;
}

// A-fragment built from an f32 row-major source (converted to f16 on load).
__device__ __forceinline__ v16h load_afrag_f32(const float* rowp, int k0, int ln) {
  const float* p = rowp + k0 + ((ln >> 4) << 3);
  v4f a0 = *(const v4f*)(p);
  v4f a1 = *(const v4f*)(p + 4);
  v4f b0 = *(const v4f*)(p + 16);
  v4f b1 = *(const v4f*)(p + 20);
  v16h r;
  #pragma unroll
  for (int i = 0; i < 4; ++i) {
    r[i]      = (_Float16)a0[i];
    r[i + 4]  = (_Float16)a1[i];
    r[i + 8]  = (_Float16)b0[i];
    r[i + 12] = (_Float16)b1[i];
  }
  return r;
}

// B-fragment (32x16, f16). colp points at the row of the *source* matrix that
// forms column N=lane&15 of B.
// lanes 0-15: elems = K0..7,K16..23 ; lanes 16-31: elems = K8..15,K24..31
__device__ __forceinline__ v16h load_bfrag_f16(const _Float16* colp, int k0, int ln) {
  const _Float16* p = colp + k0 + ((ln >> 4) << 3);
  v8h lo = *(const v8h*)(p);
  v8h hi = *(const v8h*)(p + 16);
  v16h r;
  #pragma unroll
  for (int i = 0; i < 8; ++i) { r[i] = lo[i]; r[i + 8] = hi[i]; }
  return r;
}

// ---------------- CDNA5 async copy (global -> LDS, ASYNCcnt) ----------------

__device__ __forceinline__ void async_copy_b128(_Float16* lds, const _Float16* g) {
  unsigned int      l = (unsigned int)(size_t)(void*)lds;   // LDS aperture: low 32b = LDS offset
  unsigned long long a = (unsigned long long)(size_t)(const void*)g;
  asm volatile("global_load_async_to_lds_b128 %0, %1, off" :: "v"(l), "v"(a) : "memory");
}
#define WAIT_ASYNC_LE8() asm volatile("s_wait_asynccnt 0x8" ::: "memory")
#define WAIT_ASYNC_0()   asm volatile("s_wait_asynccnt 0x0" ::: "memory")

// ---------------- Kernel 1: weight convert + zero mean output ----------------

__global__ __launch_bounds__(256) void prep_kernel(
    const float* __restrict__ Wq, const float* __restrict__ Wk,
    const float* __restrict__ Wv, const float* __restrict__ Wo,
    _Float16* __restrict__ wh, float* __restrict__ meanb) {
  int idx = blockIdx.x * 256 + threadIdx.x;
  if (idx < 4 * DIM * DIM) {
    int m = idx >> 16;              // DIM*DIM == 65536
    int e = idx & 0xFFFF;
    const float* src = (m == 0) ? Wq : (m == 1) ? Wk : (m == 2) ? Wv : Wo;
    wh[idx] = (_Float16)src[e];
  }
  if (idx < STOT) meanb[idx] = 0.0f;
}

// ---------------- Kernel 2: fused QKV projection ----------------

__global__ __launch_bounds__(256) void qkv_kernel(
    const float* __restrict__ x, const float* __restrict__ y, const float* __restrict__ z,
    const _Float16* __restrict__ wh,
    const float* __restrict__ bq, const float* __restrict__ bk, const float* __restrict__ bv,
    _Float16* __restrict__ qh, _Float16* __restrict__ kh, _Float16* __restrict__ vT) {
  int ln  = threadIdx.x & 31;
  int col = ln & 15;
  int hh  = ln >> 4;
  int wid = blockIdx.x * 8 + (threadIdx.x >> 5);
  int mat = wid / TILES;            // 0=q, 1=k, 2=v
  int t   = wid % TILES;
  int m0  = (t >> 4) * 16;          // token tile
  int n0  = (t & 15) * 16;          // feature tile

  const float*    src  = (mat == 0) ? x : (mat == 1) ? y : z;
  const _Float16* W    = wh + (size_t)mat * DIM * DIM;
  const float*    bias = (mat == 0) ? bq : (mat == 1) ? bk : bv;

  v8f acc = {};
  #pragma unroll
  for (int k0 = 0; k0 < DIM; k0 += 32) {
    v16h a = load_afrag_f32(src + (size_t)(m0 + col) * DIM, k0, ln);
    v16h b = load_bfrag_f16(W   + (size_t)(n0 + col) * DIM, k0, ln);
    acc = wmma16(a, b, acc);
  }
  float bn = bias[n0 + col];

  if (mat < 2) {
    _Float16* dst = (mat == 0) ? qh : kh;
    #pragma unroll
    for (int r = 0; r < 8; ++r) {
      int m = m0 + r + 8 * hh;
      dst[(size_t)m * DIM + n0 + col] = (_Float16)(acc[r] + bn);
    }
  } else {
    #pragma unroll
    for (int r = 0; r < 8; ++r) {
      int m  = m0 + r + 8 * hh;
      int bb = m >> 11;             // / SEQ
      int s  = m & (SEQ - 1);
      vT[((size_t)bb * DIM + n0 + col) * SEQ + s] = (_Float16)(acc[r] + bn);
    }
  }
}

// ---------------- Kernel 3: flash attention + attn column mean ----------------
// One workgroup = 8 waves sharing one K/V stream (same batch, 8 consecutive
// 16-query blocks). K/V chunks staged in LDS via async copies, double-buffered.

__global__ __launch_bounds__(256) void attn_kernel(
    const _Float16* __restrict__ qh, const _Float16* __restrict__ kh,
    const _Float16* __restrict__ vT, _Float16* __restrict__ ctxh,
    float* __restrict__ meanb) {
  __shared__ _Float16 kbuf[2][32 * DIM];   // 32 keys x 256 dims, row-major
  __shared__ _Float16 vbuf[2][DIM * 32];   // 256 dims x 32 keys, row-major

  int tid = threadIdx.x;
  int ln  = tid & 31;
  int col = ln & 15;
  int hh  = ln >> 4;
  int wid = blockIdx.x * 8 + (tid >> 5);
  int b   = wid / (SEQ / 16);
  int q0  = (wid % (SEQ / 16)) * 16;

  const _Float16* qb = qh + (size_t)b * SEQ * DIM;
  const _Float16* kb = kh + (size_t)b * SEQ * DIM;
  const _Float16* vb = vT + (size_t)b * DIM * SEQ;

  // Resident Q B-fragments: B[d][n]=q[q0+n][d], contiguous rows of q.
  v16h qf[8];
  #pragma unroll
  for (int kk = 0; kk < 8; ++kk)
    qf[kk] = load_bfrag_f16(qb + (size_t)(q0 + col) * DIM, kk * 32, ln);

  v8f ctx[16];
  #pragma unroll
  for (int i = 0; i < 16; ++i) ctx[i] = (v8f){};
  float m = -1e30f, l = 0.0f;

  // Issue chunk 0 into buffer 0: K chunk is 16KB contiguous; V chunk is 256
  // row-slices of 64B from vT. 8 async b128 per thread per chunk.
  {
    const _Float16* ks = kb + (size_t)tid * 8;
    const _Float16* vs = vb + (size_t)tid * SEQ;
    #pragma unroll
    for (int i = 0; i < 4; ++i) async_copy_b128(&kbuf[0][tid * 8 + i * 2048], ks + i * 2048);
    #pragma unroll
    for (int i = 0; i < 4; ++i) async_copy_b128(&vbuf[0][tid * 32 + i * 8], vs + i * 8);
  }

  #pragma unroll 1
  for (int kt = 0; kt < SEQ; kt += 32) {
    int cur = (kt >> 5) & 1;
    if (kt + 32 < SEQ) {
      int nxt = cur ^ 1;
      const _Float16* ks = kb + (size_t)(kt + 32) * DIM + tid * 8;
      const _Float16* vs = vb + (size_t)tid * SEQ + (kt + 32);
      #pragma unroll
      for (int i = 0; i < 4; ++i) async_copy_b128(&kbuf[nxt][tid * 8 + i * 2048], ks + i * 2048);
      #pragma unroll
      for (int i = 0; i < 4; ++i) async_copy_b128(&vbuf[nxt][tid * 32 + i * 8], vs + i * 8);
      WAIT_ASYNC_LE8();             // current chunk's 8 ops done (in-order)
    } else {
      WAIT_ASYNC_0();
    }
    __syncthreads();                // current buffers visible to all waves

    const _Float16* kl = kbuf[cur];
    const _Float16* vl = vbuf[cur];

    // S^T tiles: stA keys 0..15, stB keys 16..31 of chunk (M), queries (N)
    v8f stA = {}, stB = {};
    #pragma unroll
    for (int kk = 0; kk < 8; ++kk) {
      v16h ka = load_afrag_f16(kl + (size_t)col * DIM, kk * 32, ln);
      stA = wmma16(ka, qf[kk], stA);
    }
    #pragma unroll
    for (int kk = 0; kk < 8; ++kk) {
      v16h ka = load_afrag_f16(kl + (size_t)(16 + col) * DIM, kk * 32, ln);
      stB = wmma16(ka, qf[kk], stB);
    }

    // Online softmax; lane ln holds query q0+(ln&15), 8+8 keys in VGPRs.
    float fA[8], fB[8];
    float cm = -1e30f;
    #pragma unroll
    for (int i = 0; i < 8; ++i) {
      fA[i] = stA[i] * QKSCALE;
      fB[i] = stB[i] * QKSCALE;
      cm = fmaxf(cm, fmaxf(fA[i], fB[i]));
    }
    cm = fmaxf(cm, __shfl_xor(cm, 16, 32));
    float mn   = fmaxf(m, cm);
    float corr = __expf(m - mn);
    m = mn;
    float rs = 0.0f;
    #pragma unroll
    for (int i = 0; i < 8; ++i) {
      fA[i] = __expf(fA[i] - mn);
      fB[i] = __expf(fB[i] - mn);
      rs += fA[i] + fB[i];
    }
    rs += __shfl_xor(rs, 16, 32);
    l = l * corr + rs;

    // Rescale ctx accumulators: row r holds query r + 8*hh.
    #pragma unroll
    for (int r = 0; r < 8; ++r) {
      float cf = __shfl(corr, r + 8 * hh, 32);
      #pragma unroll
      for (int dt = 0; dt < 16; ++dt) ctx[dt][r] *= cf;
    }

    // Build P A-fragment (16 queries x 32 keys) — lane-local by construction.
    v16h pa;
    #pragma unroll
    for (int i = 0; i < 8; ++i) { pa[i] = (_Float16)fA[i]; pa[i + 8] = (_Float16)fB[i]; }

    // ctx += P @ V  (B columns = rows of vbuf, ld=32, contiguous in LDS)
    #pragma unroll
    for (int dt = 0; dt < 16; ++dt) {
      v16h vf = load_bfrag_f16(vl + (size_t)(dt * 16 + col) * 32, 0, ln);
      ctx[dt] = wmma16(pa, vf, ctx[dt]);
    }
    __syncthreads();                // everyone done reading before next overwrite
  }

  // Normalize and write ctx (f16) for the output projection.
  float invl = 1.0f / l;
  #pragma unroll
  for (int r = 0; r < 8; ++r) {
    float nf = __shfl(invl, r + 8 * hh, 32);
    int qq = q0 + r + 8 * hh;
    #pragma unroll
    for (int dt = 0; dt < 16; ++dt)
      ctxh[((size_t)b * SEQ + qq) * DIM + dt * 16 + col] = (_Float16)(ctx[dt][r] * nf);
  }

  // Pass 2: mean over queries of attn -> meanb[b*SEQ + key], via atomics.
  // Score-only recompute; K reads hit L2 (1MB/batch resident).
  const float ims = 1.0f / (float)SEQ;
  #pragma unroll 1
  for (int kt = 0; kt < SEQ; kt += 32) {
    v8f stA = {}, stB = {};
    #pragma unroll
    for (int kk = 0; kk < 8; ++kk) {
      v16h ka = load_afrag_f16(kb + (size_t)(kt + col) * DIM, kk * 32, ln);
      stA = wmma16(ka, qf[kk], stA);
      v16h kb2 = load_afrag_f16(kb + (size_t)(kt + 16 + col) * DIM, kk * 32, ln);
      stB = wmma16(kb2, qf[kk], stB);
    }
    #pragma unroll
    for (int r = 0; r < 8; ++r) {
      float pA = __expf(stA[r] * QKSCALE - m) * invl;  // attn[q=ln&15][kt+r+8*hh]
      float pB = __expf(stB[r] * QKSCALE - m) * invl;
      #pragma unroll
      for (int off = 1; off < 16; off <<= 1) {         // sum over 16 queries
        pA += __shfl_xor(pA, off, 32);
        pB += __shfl_xor(pB, off, 32);
      }
      if (col == 0) {
        atomicAdd(meanb + b * SEQ + kt + r + 8 * hh,      pA * ims);
        atomicAdd(meanb + b * SEQ + kt + 16 + r + 8 * hh, pB * ims);
      }
    }
  }
}

// ---------------- Kernel 4: output projection ----------------

__global__ __launch_bounds__(256) void outproj_kernel(
    const _Float16* __restrict__ ctxh, const _Float16* __restrict__ who,
    const float* __restrict__ bo, float* __restrict__ out) {
  int ln  = threadIdx.x & 31;
  int col = ln & 15;
  int hh  = ln >> 4;
  int wid = blockIdx.x * 8 + (threadIdx.x >> 5);
  int m0  = (wid >> 4) * 16;
  int n0  = (wid & 15) * 16;

  v8f acc = {};
  #pragma unroll
  for (int k0 = 0; k0 < DIM; k0 += 32) {
    v16h a = load_afrag_f16(ctxh + (size_t)(m0 + col) * DIM, k0, ln);
    v16h b = load_bfrag_f16(who  + (size_t)(n0 + col) * DIM, k0, ln);
    acc = wmma16(a, b, acc);
  }
  float bn = bo[n0 + col];
  #pragma unroll
  for (int r = 0; r < 8; ++r)
    out[(size_t)(m0 + r + 8 * hh) * DIM + n0 + col] = acc[r] + bn;
}

// ---------------- launch ----------------

extern "C" void kernel_launch(void* const* d_in, const int* in_sizes, int n_in,
                              void* d_out, int out_size, void* d_ws, size_t ws_size,
                              hipStream_t stream) {
  const float* x  = (const float*)d_in[0];
  const float* y  = (const float*)d_in[1];
  const float* z  = (const float*)d_in[2];
  const float* Wq = (const float*)d_in[3];
  const float* bq = (const float*)d_in[4];
  const float* Wk = (const float*)d_in[5];
  const float* bk = (const float*)d_in[6];
  const float* Wv = (const float*)d_in[7];
  const float* bv = (const float*)d_in[8];
  const float* Wo = (const float*)d_in[9];
  const float* bo = (const float*)d_in[10];

  float* out   = (float*)d_out;                 // [STOT*DIM] f32
  float* meanb = out + (size_t)STOT * DIM;      // [STOT] f32 (second output)

  char* ws = (char*)d_ws;
  _Float16* wh   = (_Float16*)ws;                          // 4 * DIM*DIM f16
  _Float16* qh   = (_Float16*)(ws + (size_t)4 * DIM * DIM * 2);
  _Float16* kh   = qh + (size_t)STOT * DIM;
  _Float16* vT   = kh + (size_t)STOT * DIM;                // [B][D][S]
  _Float16* ctxh = vT + (size_t)STOT * DIM;
  _Float16* who  = wh + (size_t)3 * DIM * DIM;

  prep_kernel<<<(4 * DIM * DIM + 255) / 256, 256, 0, stream>>>(Wq, Wk, Wv, Wo, wh, meanb);
  qkv_kernel<<<(3 * TILES) / 8, 256, 0, stream>>>(x, y, z, wh, bq, bk, bv, qh, kh, vT);
  attn_kernel<<<(BATCH * (SEQ / 16)) / 8, 256, 0, stream>>>(qh, kh, vT, ctxh, meanb);
  outproj_kernel<<<TILES / 8, 256, 0, stream>>>(ctxh, who, bo, out);
}
